// SEGNN_23862838297392
// MI455X (gfx1250) — matrix operand order repaired and checked by reading, not compiled, not run on hardware
//
#include <hip/hip_runtime.h>
#include <hip/hip_bf16.h>
#include <math.h>

// ---------------------------------------------------------------------------
// SEGNN forward for MI455X (gfx1250, wave32).
// All dense TP paths run on V_WMMA_F32_16X16X32_BF16 (bf16 A/B, f32 C/D).
// 16-row tiles (edges/nodes) per wave; activations & weights staged in LDS.
// This revision: native bf16 converts + packed b32 LDS staging stores +
// vectorized global gathers (halves staging VALU/DS vs round 1).
// ---------------------------------------------------------------------------

typedef __attribute__((ext_vector_type(8)))  float  v8f;
typedef __attribute__((ext_vector_type(8)))  __bf16 v8bf;
typedef __attribute__((ext_vector_type(16))) __bf16 v16bf;
typedef unsigned short u16;
typedef unsigned int u32;

// ---------------- LDS layouts (u16 units) ----------------------------------
// Edge-message kernel (msg0 + msg1 fused)
constexpr int E_W0SS = 0;              // 96x64
constexpr int E_W0VS = 6144;           // 64x64
constexpr int E_W0SV = 10240;          // 96x32
constexpr int E_W0VVC = 13312;         // 128x32 ([vv;vc])
constexpr int E_W1SS = 17408;          // 32x64
constexpr int E_W1VS = 19456;          // 32x64
constexpr int E_W1SV = 21504;          // 32x32
constexpr int E_W1VVC = 22528;         // 64x32
constexpr int E_WEND = 24576;
constexpr int EO_BUFS = 0, EO_BUFD = 1536, EO_BUFV = 2560;
constexpr int EO_BUFS1 = 8704, EO_BUFD1 = 9216, EO_BUFV1 = 9728;
constexpr int EO_ATTR = 12800, EO_IRECV = 12896;
constexpr int E_PW = 12928;
constexpr size_t E_LDS = (size_t)(E_WEND + 4 * E_PW) * 2;   // 152576 B

// Node-update kernel (upd0 + upd1 fused)
constexpr int N_U0SS = 0;              // 64x64
constexpr int N_U0VS = 4096;           // 64x64
constexpr int N_U0SV = 8192;           // 64x32
constexpr int N_U0VVC = 10240;         // 128x32
constexpr int N_U1SS = 14336;          // 32x32
constexpr int N_U1VS = 15360;          // 32x32
constexpr int N_U1SV = 16384;          // 32x32
constexpr int N_U1VVC = 17408;         // 64x32
constexpr int N_WEND = 19456;
constexpr int NO_BUFS = 0, NO_BUFD = 1024, NO_BUFV = 2048;
constexpr int NO_BUFS1 = 8192, NO_BUFD1 = 8704, NO_BUFV1 = 9216;
constexpr int NO_ATTR = 12288;
constexpr int N_PW = 12416;
constexpr size_t N_LDS = (size_t)(N_WEND + 4 * N_PW) * 2;   // 138240 B

// Prepool kernel (prepool0 gate + prepool1 scalar TP)
constexpr int P_P0SS = 0;              // 32x64
constexpr int P_P0VS = 2048;           // 32x64
constexpr int P_P0SV = 4096;           // 32x32
constexpr int P_P0VVC = 5120;          // 64x32
constexpr int P_P1SS = 7168;           // 32x64
constexpr int P_P1VS = 9216;           // 32x64
constexpr int P_WEND = 11264;
constexpr int PO_BUFS = 0, PO_BUFD = 512, PO_BUFV = 1024;
constexpr int PO_BUFS1 = 4096, PO_BUFD1 = 4608, PO_BUFV1 = 5120;
constexpr int PO_ATTR = 8192, PO_IG = 8288;
constexpr int P_PW = 8352;
constexpr size_t P_LDS = (size_t)(P_WEND + 4 * P_PW) * 2;   // 89344 B

// ---------------- small helpers --------------------------------------------
// Native f32 -> bf16 convert (lowers to hardware cvt; RNE).
__device__ __forceinline__ u16 f2bf(float f) {
  union { __bf16 h; u16 u; } c;
  c.h = (__bf16)f;
  return c.u;
}
// Pack two bf16 into one dword (compiler can match v_cvt_pk_bf16_f32).
__device__ __forceinline__ u32 pk2(float lo, float hi) {
  return (u32)f2bf(lo) | ((u32)f2bf(hi) << 16);
}
// 4B-aligned packed LDS store of two bf16.
__device__ __forceinline__ void st2(u16* p, float lo, float hi) {
  *reinterpret_cast<u32*>(p) = pk2(lo, hi);
}

__device__ __forceinline__ v8f zero8() {
  v8f z;
#pragma unroll
  for (int i = 0; i < 8; ++i) z[i] = 0.f;
  return z;
}

__device__ __forceinline__ v16bf cat16(v8bf lo, v8bf hi) {
  return __builtin_shufflevector(lo, hi, 0, 1, 2, 3, 4, 5, 6, 7,
                                 8, 9, 10, 11, 12, 13, 14, 15);
}

// A-fragment (16x32 bf16): lane l -> row m=l&15, khalf=(l>>4); two contiguous
// 8-element runs at K = khalf*8 and K = 16 + khalf*8 (ISA 7.12.2).
__device__ __forceinline__ v16bf loadA(const u16* tile, int K, int kt, int lane) {
  int m = lane & 15, kh = (lane >> 4) & 1;
  const u16* row = tile + m * K + kt * 32 + kh * 8;
  v8bf lo = *reinterpret_cast<const v8bf*>(row);
  v8bf hi = *reinterpret_cast<const v8bf*>(row + 16);
  return cat16(lo, hi);
}

// B-fragment (32x16 bf16): lane l holds column N=l&15, khalf=(l>>4); K runs at
// khalf*8 and 16 + khalf*8 (same order as A).
__device__ __forceinline__ v16bf loadB(const u16* W, int Nout, int kt, int nt, int lane) {
  int n = lane & 15, kh = (lane >> 4) & 1;
  const __bf16* p = reinterpret_cast<const __bf16*>(W) + (kt * 32 + kh * 8) * Nout + nt * 16 + n;
  v16bf b;
#pragma unroll
  for (int s = 0; s < 16; ++s) b[s] = p[(s + (s >= 8 ? 8 : 0)) * Nout];
  return b;
}

__device__ __forceinline__ v8f wmma_b(v16bf a, v16bf b, v8f c) {
  return __builtin_amdgcn_wmma_f32_16x16x32_bf16(false, a, false, b, (short)0, c,
                                                 false, false);
}

__device__ __forceinline__ float sig_(float x) { return 1.f / (1.f + __expf(-x)); }
__device__ __forceinline__ float silu_(float x) { return x * sig_(x); }

// Weight staging: f32 global -> bf16 LDS, packed pair stores.
// (all nvalid/ntot are even; all dst offsets are even u16 -> 4B aligned)
__device__ __forceinline__ void loadW1(u16* dst, const float* src, int nvalid,
                                       int ntot, int tid, int nth) {
  for (int i = 2 * tid; i < ntot; i += 2 * nth) {
    float lo = (i < nvalid) ? src[i] : 0.f;
    float hi = (i + 1 < nvalid) ? src[i + 1] : 0.f;
    st2(dst + i, lo, hi);
  }
}
__device__ __forceinline__ void loadW2(u16* dst, const float* a, const float* b,
                                       int na, int tid, int nth) {
  for (int i = 2 * tid; i < 2 * na; i += 2 * nth) {
    float lo = (i < na) ? a[i] : b[i - na];
    float hi = (i + 1 < na) ? a[i + 1] : b[i + 1 - na];
    st2(dst + i, lo, hi);
  }
}

__device__ __forceinline__ void load_a1reg(const float* attr1, int lane,
                                           float (&a1reg)[8][3]) {
  int m0 = 8 * ((lane >> 4) & 1);
#pragma unroll
  for (int r = 0; r < 8; ++r)
#pragma unroll
    for (int c = 0; c < 3; ++c) a1reg[r][c] = attr1[(m0 + r) * 3 + c];
}

// ---------------- generic fused tensor product -----------------------------
// s_all = (bufS @ Wss + bufD @ Wvs) * inv_s                        [16 x NS]
// v_c   = ((bufS @ Wsv) * attr1_c + bufV_c @ Wvvc) * inv_v         [16 x 32]
// (a0/e0 == 1 by construction in setup_inputs; folded out.)
template <int KS, int KD, int KV, int NS>
__device__ __forceinline__ void tp_compute(const u16* bufS, const u16* bufD,
                                           const u16* bufV, const u16* Wss,
                                           const u16* Wvs, const u16* Wsv,
                                           const u16* Wvvc,
                                           const float (&a1reg)[8][3], float inv_s,
                                           float inv_v, int lane,
                                           v8f (&sOut)[NS / 16], v8f (&vOut)[3][2]) {
#pragma unroll
  for (int nt = 0; nt < NS / 16; ++nt) {
    v8f acc = zero8();
#pragma unroll
    for (int kt = 0; kt < KS / 32; ++kt)
      acc = wmma_b(loadA(bufS, KS, kt, lane), loadB(Wss, NS, kt, nt, lane), acc);
#pragma unroll
    for (int kt = 0; kt < KD / 32; ++kt)
      acc = wmma_b(loadA(bufD, KD, kt, lane), loadB(Wvs, NS, kt, nt, lane), acc);
#pragma unroll
    for (int r = 0; r < 8; ++r) acc[r] *= inv_s;
    sOut[nt] = acc;
  }
  v8f msv[2];
#pragma unroll
  for (int nt = 0; nt < 2; ++nt) {
    v8f acc = zero8();
#pragma unroll
    for (int kt = 0; kt < KS / 32; ++kt)
      acc = wmma_b(loadA(bufS, KS, kt, lane), loadB(Wsv, 32, kt, nt, lane), acc);
    msv[nt] = acc;
  }
#pragma unroll
  for (int c = 0; c < 3; ++c) {
#pragma unroll
    for (int nt = 0; nt < 2; ++nt) {
      v8f acc = zero8();
#pragma unroll
      for (int kt = 0; kt < KV / 32; ++kt)
        acc = wmma_b(loadA(bufV + c * 16 * KV, KV, kt, lane),
                     loadB(Wvvc, 32, kt, nt, lane), acc);
      v8f o;
#pragma unroll
      for (int r = 0; r < 8; ++r) o[r] = (msv[nt][r] * a1reg[r][c] + acc[r]) * inv_v;
      vOut[c][nt] = o;
    }
  }
}

// scalar-only TP (prepool1)
template <int KS, int KD, int NS>
__device__ __forceinline__ void tp_scalar(const u16* bufS, const u16* bufD,
                                          const u16* Wss, const u16* Wvs,
                                          float inv_s, int lane,
                                          v8f (&sOut)[NS / 16]) {
#pragma unroll
  for (int nt = 0; nt < NS / 16; ++nt) {
    v8f acc = zero8();
#pragma unroll
    for (int kt = 0; kt < KS / 32; ++kt)
      acc = wmma_b(loadA(bufS, KS, kt, lane), loadB(Wss, NS, kt, nt, lane), acc);
#pragma unroll
    for (int kt = 0; kt < KD / 32; ++kt)
      acc = wmma_b(loadA(bufD, KD, kt, lane), loadB(Wvs, NS, kt, nt, lane), acc);
#pragma unroll
    for (int r = 0; r < 8; ++r) acc[r] *= inv_s;
    sOut[nt] = acc;
  }
}

// gate (silu/sigmoid) + restage as msg1/upd1/prepool1 inputs
// bufS1 [16][32], bufD1 [16][32], bufV1 [3][16][64] = [mv_c | cross_c]
__device__ __forceinline__ void gate_restage(const v8f (&sA)[4], const v8f (&vA)[3][2],
                                             const float (&a1reg)[8][3], u16* bufS1,
                                             u16* bufD1, u16* bufV1, int lane) {
  int m0 = 8 * ((lane >> 4) & 1), n0 = lane & 15;
#pragma unroll
  for (int t = 0; t < 2; ++t) {
#pragma unroll
    for (int r = 0; r < 8; ++r) {
      int M = m0 + r, n = t * 16 + n0;
      bufS1[M * 32 + n] = f2bf(silu_(sA[t][r]));
      float g = sig_(sA[2 + t][r]);
      float mx = vA[0][t][r] * g, my = vA[1][t][r] * g, mz = vA[2][t][r] * g;
      float ax = a1reg[r][0], ay = a1reg[r][1], az = a1reg[r][2];
      bufD1[M * 32 + n] = f2bf(mx * ax + my * ay + mz * az);
      float cx = my * az - mz * ay, cy = mz * ax - mx * az, cz = mx * ay - my * ax;
      bufV1[0 * 1024 + M * 64 + n] = f2bf(mx);
      bufV1[1 * 1024 + M * 64 + n] = f2bf(my);
      bufV1[2 * 1024 + M * 64 + n] = f2bf(mz);
      bufV1[0 * 1024 + M * 64 + 32 + n] = f2bf(cx);
      bufV1[1 * 1024 + M * 64 + 32 + n] = f2bf(cy);
      bufV1[2 * 1024 + M * 64 + 32 + n] = f2bf(cz);
    }
  }
}

// ---------------- staging from global --------------------------------------
// Stage one node/sender-receiver row pair of vector channels (2 at a time).
// base cols: mv at vcol+k, cross at vcol+KV/2+k (KV = 128 edge/node, 64 pre).
template <int KV>
__device__ __forceinline__ void stage_vpair(const float* vp, float ex, float ey,
                                            float ez, u16* bufD, int dcol,
                                            u16* bufV, int j, int vcol) {
  // vp: 6 contiguous floats = channels k, k+1
  float a0 = vp[0], a1 = vp[1], a2 = vp[2], b0 = vp[3], b1 = vp[4], b2 = vp[5];
  st2(bufD + dcol, a0 * ex + a1 * ey + a2 * ez, b0 * ex + b1 * ey + b2 * ez);
  u16* row0 = bufV + 0 * 16 * KV + j * KV + vcol;
  u16* row1 = bufV + 1 * 16 * KV + j * KV + vcol;
  u16* row2 = bufV + 2 * 16 * KV + j * KV + vcol;
  st2(row0, a0, b0);
  st2(row1, a1, b1);
  st2(row2, a2, b2);
  st2(row0 + KV / 2, a1 * ez - a2 * ey, b1 * ez - b2 * ey);
  st2(row1 + KV / 2, a2 * ex - a0 * ez, b2 * ex - b0 * ez);
  st2(row2 + KV / 2, a0 * ey - a1 * ex, b0 * ey - b1 * ex);
}

__device__ __forceinline__ void stage_edge(int e, int lane, const float* sg,
                                           const float* vg, const float* e1g,
                                           const float* feg, const int* snd,
                                           const int* rcv, u16* bufS, u16* bufD,
                                           u16* bufV, float* attr1, int* irecv) {
  int j = lane >> 1, h = lane & 1;
  int is = snd[e], ir = rcv[e];
  float ex = e1g[e * 3 + 0], ey = e1g[e * 3 + 1], ez = e1g[e * 3 + 2];
  if (h == 0) {
    attr1[j * 3 + 0] = ex; attr1[j * 3 + 1] = ey; attr1[j * 3 + 2] = ez;
    irecv[j] = ir;
    st2(&bufS[j * 96 + 64], feg[e * 2 + 0], feg[e * 2 + 1]);
  } else {
    // zero K-pad 66..95 (15 packed pair stores)
#pragma unroll
    for (int t = 0; t < 15; ++t) st2(&bufS[j * 96 + 66 + 2 * t], 0.f, 0.f);
  }
  const float4* ssv = reinterpret_cast<const float4*>(sg + (size_t)is * 32 + h * 16);
  const float4* srv = reinterpret_cast<const float4*>(sg + (size_t)ir * 32 + h * 16);
#pragma unroll
  for (int i = 0; i < 4; ++i) {
    float4 a = ssv[i], b = srv[i];
    st2(&bufS[j * 96 + h * 16 + i * 4 + 0], a.x, a.y);
    st2(&bufS[j * 96 + h * 16 + i * 4 + 2], a.z, a.w);
    st2(&bufS[j * 96 + 32 + h * 16 + i * 4 + 0], b.x, b.y);
    st2(&bufS[j * 96 + 32 + h * 16 + i * 4 + 2], b.z, b.w);
  }
#pragma unroll
  for (int q = 0; q < 16; q += 2) {
    int k = h * 16 + q;
    stage_vpair<128>(vg + ((size_t)is * 32 + k) * 3, ex, ey, ez, bufD, j * 64 + k,
                     bufV, j, k);
    stage_vpair<128>(vg + ((size_t)ir * 32 + k) * 3, ex, ey, ez, bufD,
                     j * 64 + 32 + k, bufV, j, 32 + k);
  }
}

__device__ __forceinline__ void stage_node(int nd, int lane, const float* sg,
                                           const float* vg, const float* ags,
                                           const float* agv, const float* a1g,
                                           u16* bufS, u16* bufD, u16* bufV,
                                           float* attr1) {
  int j = lane >> 1, h = lane & 1;
  float ax = a1g[nd * 3 + 0], ay = a1g[nd * 3 + 1], az = a1g[nd * 3 + 2];
  if (h == 0) { attr1[j * 3] = ax; attr1[j * 3 + 1] = ay; attr1[j * 3 + 2] = az; }
  const float4* s1 = reinterpret_cast<const float4*>(sg + (size_t)nd * 32 + h * 16);
  const float4* s2 = reinterpret_cast<const float4*>(ags + (size_t)nd * 32 + h * 16);
#pragma unroll
  for (int i = 0; i < 4; ++i) {
    float4 a = s1[i], b = s2[i];
    st2(&bufS[j * 64 + h * 16 + i * 4 + 0], a.x, a.y);
    st2(&bufS[j * 64 + h * 16 + i * 4 + 2], a.z, a.w);
    st2(&bufS[j * 64 + 32 + h * 16 + i * 4 + 0], b.x, b.y);
    st2(&bufS[j * 64 + 32 + h * 16 + i * 4 + 2], b.z, b.w);
  }
#pragma unroll
  for (int q = 0; q < 16; q += 2) {
    int k = h * 16 + q;
    stage_vpair<128>(vg + ((size_t)nd * 32 + k) * 3, ax, ay, az, bufD, j * 64 + k,
                     bufV, j, k);
    stage_vpair<128>(agv + ((size_t)nd * 32 + k) * 3, ax, ay, az, bufD,
                     j * 64 + 32 + k, bufV, j, 32 + k);
  }
}

__device__ __forceinline__ void stage_pre(int nd, int lane, const float* sg,
                                          const float* vg, const float* a1g,
                                          const int* gidx, u16* bufS, u16* bufD,
                                          u16* bufV, float* attr1, int* igr) {
  int j = lane >> 1, h = lane & 1;
  float ax = a1g[nd * 3 + 0], ay = a1g[nd * 3 + 1], az = a1g[nd * 3 + 2];
  if (h == 0) {
    attr1[j * 3] = ax; attr1[j * 3 + 1] = ay; attr1[j * 3 + 2] = az;
    igr[j] = gidx[nd];
  }
  const float4* s1 = reinterpret_cast<const float4*>(sg + (size_t)nd * 32 + h * 16);
#pragma unroll
  for (int i = 0; i < 4; ++i) {
    float4 a = s1[i];
    st2(&bufS[j * 32 + h * 16 + i * 4 + 0], a.x, a.y);
    st2(&bufS[j * 32 + h * 16 + i * 4 + 2], a.z, a.w);
  }
#pragma unroll
  for (int q = 0; q < 16; q += 2) {
    int k = h * 16 + q;
    stage_vpair<64>(vg + ((size_t)nd * 32 + k) * 3, ax, ay, az, bufD, j * 32 + k,
                    bufV, j, k);
  }
}

// ---------------- kernels ---------------------------------------------------
__global__ void k_embed(const float* sin, const float* vin, const float* a1g,
                        const float* wss, const float* wvs, const float* wsv,
                        const float* wvv, const float* wvc, float* sg, float* vg,
                        int N) {
  int i = blockIdx.x * blockDim.x + threadIdx.x;
  if (i >= N) return;
  float s0 = sin[i * 2], s1 = sin[i * 2 + 1];
  float ax = a1g[i * 3], ay = a1g[i * 3 + 1], az = a1g[i * 3 + 2];
  float v0x = vin[i * 6 + 0], v0y = vin[i * 6 + 1], v0z = vin[i * 6 + 2];
  float v1x = vin[i * 6 + 3], v1y = vin[i * 6 + 4], v1z = vin[i * 6 + 5];
  float d0 = v0x * ax + v0y * ay + v0z * az, d1 = v1x * ax + v1y * ay + v1z * az;
  float c0x = v0y * az - v0z * ay, c0y = v0z * ax - v0x * az, c0z = v0x * ay - v0y * ax;
  float c1x = v1y * az - v1z * ay, c1y = v1z * ax - v1x * az, c1z = v1x * ay - v1y * ax;
  const float rs6 = 0.40824829046386302f;  // 1/sqrt(6)
#pragma unroll 4
  for (int o = 0; o < 32; ++o) {
    sg[(size_t)i * 32 + o] =
        (s0 * wss[o] + s1 * wss[32 + o] + d0 * wvs[o] + d1 * wvs[32 + o]) * 0.5f;
    float msv = s0 * wsv[o] + s1 * wsv[32 + o];
    float w0 = wvv[o], w1 = wvv[32 + o], q0 = wvc[o], q1 = wvc[32 + o];
    float* vp = vg + ((size_t)i * 32 + o) * 3;
    vp[0] = (msv * ax + v0x * w0 + v1x * w1 + c0x * q0 + c1x * q1) * rs6;
    vp[1] = (msv * ay + v0y * w0 + v1y * w1 + c0y * q0 + c1y * q1) * rs6;
    vp[2] = (msv * az + v0z * w0 + v1z * w1 + c0z * q0 + c1z * q1) * rs6;
  }
}

__global__ void __launch_bounds__(128) k_edge(
    const float* sg, const float* vg, const float* e1g, const float* feg,
    const int* snd, const int* rcv, float* aggS, float* aggV, const float* w_ss0,
    const float* w_vs0, const float* w_sv0, const float* w_vv0, const float* w_vc0,
    const float* w_ss1, const float* w_vs1, const float* w_sv1, const float* w_vv1,
    const float* w_vc1, int numTiles, int E) {
  extern __shared__ u16 smem[];
  int tid = threadIdx.x, lane = tid & 31, wv = tid >> 5, nth = blockDim.x;
  loadW1(smem + E_W0SS, w_ss0, 66 * 64, 96 * 64, tid, nth);
  loadW1(smem + E_W0VS, w_vs0, 64 * 64, 64 * 64, tid, nth);
  loadW1(smem + E_W0SV, w_sv0, 66 * 32, 96 * 32, tid, nth);
  loadW2(smem + E_W0VVC, w_vv0, w_vc0, 64 * 32, tid, nth);
  loadW1(smem + E_W1SS, w_ss1, 32 * 64, 32 * 64, tid, nth);
  loadW1(smem + E_W1VS, w_vs1, 32 * 64, 32 * 64, tid, nth);
  loadW1(smem + E_W1SV, w_sv1, 32 * 32, 32 * 32, tid, nth);
  loadW2(smem + E_W1VVC, w_vv1, w_vc1, 32 * 32, tid, nth);
  u16* pw = smem + E_WEND + wv * E_PW;
  float* attr1 = (float*)(pw + EO_ATTR);
  int* irecv = (int*)(pw + EO_IRECV);
  int tile = blockIdx.x * 4 + wv;
  bool act = tile < numTiles;
  int e = tile * 16 + (lane >> 1);
  if (act && e < E)
    stage_edge(e, lane, sg, vg, e1g, feg, snd, rcv, pw + EO_BUFS, pw + EO_BUFD,
               pw + EO_BUFV, attr1, irecv);
  __syncthreads();
  float a1reg[8][3];
  v8f sA[4], vA[3][2];
  if (act) {
    load_a1reg(attr1, lane, a1reg);
    tp_compute<96, 64, 128, 64>(pw + EO_BUFS, pw + EO_BUFD, pw + EO_BUFV,
                                smem + E_W0SS, smem + E_W0VS, smem + E_W0SV,
                                smem + E_W0VVC, a1reg, rsqrtf(130.f), rsqrtf(194.f),
                                lane, sA, vA);
    gate_restage(sA, vA, a1reg, pw + EO_BUFS1, pw + EO_BUFD1, pw + EO_BUFV1, lane);
  }
  __syncthreads();
  if (act) {
    tp_compute<32, 32, 64, 64>(pw + EO_BUFS1, pw + EO_BUFD1, pw + EO_BUFV1,
                               smem + E_W1SS, smem + E_W1VS, smem + E_W1SV,
                               smem + E_W1VVC, a1reg, rsqrtf(64.f), rsqrtf(96.f),
                               lane, sA, vA);
    int m0 = 8 * ((lane >> 4) & 1), n0 = lane & 15;
#pragma unroll
    for (int t = 0; t < 2; ++t) {
#pragma unroll
      for (int r = 0; r < 8; ++r) {
        int M = m0 + r;
        if (tile * 16 + M < E) {
          int n = t * 16 + n0, dst = irecv[M];
          atomicAdd(&aggS[(size_t)dst * 32 + n], silu_(sA[t][r]));
          float g = sig_(sA[2 + t][r]);
          float* vp = &aggV[((size_t)dst * 32 + n) * 3];
          atomicAdd(vp + 0, vA[0][t][r] * g);
          atomicAdd(vp + 1, vA[1][t][r] * g);
          atomicAdd(vp + 2, vA[2][t][r] * g);
        }
      }
    }
  }
}

__global__ void __launch_bounds__(128) k_node(
    float* sg, float* vg, const float* ags, const float* agv, const float* a1g,
    const float* u0ss, const float* u0vs, const float* u0sv, const float* u0vv,
    const float* u0vc, const float* u1ss, const float* u1vs, const float* u1sv,
    const float* u1vv, const float* u1vc, int numTiles, int N) {
  extern __shared__ u16 smem[];
  int tid = threadIdx.x, lane = tid & 31, wv = tid >> 5, nth = blockDim.x;
  loadW1(smem + N_U0SS, u0ss, 64 * 64, 64 * 64, tid, nth);
  loadW1(smem + N_U0VS, u0vs, 64 * 64, 64 * 64, tid, nth);
  loadW1(smem + N_U0SV, u0sv, 64 * 32, 64 * 32, tid, nth);
  loadW2(smem + N_U0VVC, u0vv, u0vc, 64 * 32, tid, nth);
  loadW1(smem + N_U1SS, u1ss, 32 * 32, 32 * 32, tid, nth);
  loadW1(smem + N_U1VS, u1vs, 32 * 32, 32 * 32, tid, nth);
  loadW1(smem + N_U1SV, u1sv, 32 * 32, 32 * 32, tid, nth);
  loadW2(smem + N_U1VVC, u1vv, u1vc, 32 * 32, tid, nth);
  u16* pw = smem + N_WEND + wv * N_PW;
  float* attr1 = (float*)(pw + NO_ATTR);
  int tile = blockIdx.x * 4 + wv;
  bool act = tile < numTiles;
  int nd = tile * 16 + (lane >> 1);
  if (act && nd < N)
    stage_node(nd, lane, sg, vg, ags, agv, a1g, pw + NO_BUFS, pw + NO_BUFD,
               pw + NO_BUFV, attr1);
  __syncthreads();
  float a1reg[8][3];
  v8f sA[4], vA[3][2];
  if (act) {
    load_a1reg(attr1, lane, a1reg);
    tp_compute<64, 64, 128, 64>(pw + NO_BUFS, pw + NO_BUFD, pw + NO_BUFV,
                                smem + N_U0SS, smem + N_U0VS, smem + N_U0SV,
                                smem + N_U0VVC, a1reg, rsqrtf(128.f), rsqrtf(192.f),
                                lane, sA, vA);
    gate_restage(sA, vA, a1reg, pw + NO_BUFS1, pw + NO_BUFD1, pw + NO_BUFV1, lane);
  }
  __syncthreads();
  if (act) {
    v8f dsA[2];
    tp_compute<32, 32, 64, 32>(pw + NO_BUFS1, pw + NO_BUFD1, pw + NO_BUFV1,
                               smem + N_U1SS, smem + N_U1VS, smem + N_U1SV,
                               smem + N_U1VVC, a1reg, rsqrtf(64.f), rsqrtf(96.f),
                               lane, dsA, vA);
    int m0 = 8 * ((lane >> 4) & 1), n0 = lane & 15;
#pragma unroll
    for (int t = 0; t < 2; ++t) {
#pragma unroll
      for (int r = 0; r < 8; ++r) {
        int M = m0 + r, node = tile * 16 + M;
        if (node < N) {
          int n = t * 16 + n0;
          sg[(size_t)node * 32 + n] += dsA[t][r];  // residual
          float* vp = &vg[((size_t)node * 32 + n) * 3];
          vp[0] += vA[0][t][r];
          vp[1] += vA[1][t][r];
          vp[2] += vA[2][t][r];
        }
      }
    }
  }
}

__global__ void __launch_bounds__(128) k_pre(
    const float* sg, const float* vg, const float* a1g, const int* gidx,
    float* pooled, const float* p0ss, const float* p0vs, const float* p0sv,
    const float* p0vv, const float* p0vc, const float* p1ss, const float* p1vs,
    int numTiles, int N) {
  extern __shared__ u16 smem[];
  int tid = threadIdx.x, lane = tid & 31, wv = tid >> 5, nth = blockDim.x;
  loadW1(smem + P_P0SS, p0ss, 32 * 64, 32 * 64, tid, nth);
  loadW1(smem + P_P0VS, p0vs, 32 * 64, 32 * 64, tid, nth);
  loadW1(smem + P_P0SV, p0sv, 32 * 32, 32 * 32, tid, nth);
  loadW2(smem + P_P0VVC, p0vv, p0vc, 32 * 32, tid, nth);
  loadW1(smem + P_P1SS, p1ss, 32 * 64, 32 * 64, tid, nth);
  loadW1(smem + P_P1VS, p1vs, 32 * 64, 32 * 64, tid, nth);
  u16* pw = smem + P_WEND + wv * P_PW;
  float* attr1 = (float*)(pw + PO_ATTR);
  int* igr = (int*)(pw + PO_IG);
  int tile = blockIdx.x * 4 + wv;
  bool act = tile < numTiles;
  int nd = tile * 16 + (lane >> 1);
  if (act && nd < N)
    stage_pre(nd, lane, sg, vg, a1g, gidx, pw + PO_BUFS, pw + PO_BUFD, pw + PO_BUFV,
              attr1, igr);
  __syncthreads();
  float a1reg[8][3];
  v8f sA[4], vA[3][2];
  if (act) {
    load_a1reg(attr1, lane, a1reg);
    tp_compute<32, 32, 64, 64>(pw + PO_BUFS, pw + PO_BUFD, pw + PO_BUFV,
                               smem + P_P0SS, smem + P_P0VS, smem + P_P0SV,
                               smem + P_P0VVC, a1reg, rsqrtf(64.f), rsqrtf(96.f),
                               lane, sA, vA);
    gate_restage(sA, vA, a1reg, pw + PO_BUFS1, pw + PO_BUFD1, pw + PO_BUFV1, lane);
  }
  __syncthreads();
  if (act) {
    v8f sP[4];
    tp_scalar<32, 32, 64>(pw + PO_BUFS1, pw + PO_BUFD1, smem + P_P1SS,
                          smem + P_P1VS, 0.125f, lane, sP);
    int m0 = 8 * ((lane >> 4) & 1), n0 = lane & 15;
#pragma unroll
    for (int t = 0; t < 4; ++t) {
#pragma unroll
      for (int r = 0; r < 8; ++r) {
        int M = m0 + r, node = tile * 16 + M;
        if (node < N) atomicAdd(&pooled[(size_t)igr[M] * 64 + t * 16 + n0], sP[t][r]);
      }
    }
  }
}

__global__ void k_counts(const int* gidx, float* counts, int N) {
  int i = blockIdx.x * blockDim.x + threadIdx.x;
  if (i < N) atomicAdd(&counts[gidx[i]], 1.0f);
}

__global__ void k_post(const float* pooled, const float* counts, const float* wpp,
                       const float* wout, float* out, int G) {
  int g = blockIdx.x * blockDim.x + threadIdx.x;
  if (g >= G) return;
  float cnt = counts[g];
  cnt = cnt > 1.f ? cnt : 1.f;
  float p[64];
#pragma unroll
  for (int k = 0; k < 64; ++k) p[k] = pooled[(size_t)g * 64 + k] / cnt;
  float o = 0.f;
  for (int j = 0; j < 64; ++j) {
    float h = 0.f;
#pragma unroll
    for (int k = 0; k < 64; ++k) h += p[k] * wpp[k * 64 + j];
    h *= 0.125f;                 // 1/sqrt(64)
    o += (h / (1.f + __expf(-h))) * wout[j];
  }
  out[g] = o * 0.125f;           // output fan 1/sqrt(64)
}

// ---------------- launcher ---------------------------------------------------
extern "C" void kernel_launch(void* const* d_in, const int* in_sizes, int n_in,
                              void* d_out, int out_size, void* d_ws, size_t ws_size,
                              hipStream_t stream) {
  (void)n_in; (void)ws_size;
  const int N = in_sizes[0] / 2;   // s_nodes [N,2]
  const int E = in_sizes[7];       // senders [E]
  const int G = out_size;
  auto PF = [&](int i) { return (const float*)d_in[i]; };
  auto PI = [&](int i) { return (const int*)d_in[i]; };

  float* ws = (float*)d_ws;
  float* Sb = ws;                              // [N,32]
  float* Vb = Sb + (size_t)N * 32;             // [N,32,3]
  float* AggS = Vb + (size_t)N * 96;           // [N,32]
  float* AggV = AggS + (size_t)N * 32;         // [N,32,3]
  float* Pooled = AggV + (size_t)N * 96;       // [G,64]
  float* Counts = Pooled + (size_t)G * 64;     // [G]

  hipFuncSetAttribute((const void*)k_edge, hipFuncAttributeMaxDynamicSharedMemorySize,
                      (int)E_LDS);
  hipFuncSetAttribute((const void*)k_node, hipFuncAttributeMaxDynamicSharedMemorySize,
                      (int)N_LDS);
  hipFuncSetAttribute((const void*)k_pre, hipFuncAttributeMaxDynamicSharedMemorySize,
                      (int)P_LDS);

  // d_in layout: 0..9 data tensors, then params in setup_inputs insertion order:
  // 10..14 embed{ss,vs,sv,vv,vc}; per layer (20): msg0{5} msg1{5} upd0{5} upd1{5};
  // 55..59 prepool0; 60..61 prepool1{ss,vs}; 62 postpool0.ss; 63 output.ss
  k_embed<<<(N + 255) / 256, 256, 0, stream>>>(PF(0), PF(1), PF(3), PF(10), PF(11),
                                               PF(12), PF(13), PF(14), Sb, Vb, N);

  const int tE = (E + 15) / 16, tN = (N + 15) / 16;
  for (int L = 0; L < 2; ++L) {
    int b = 15 + L * 20;
    hipMemsetAsync(AggS, 0, (size_t)N * 128 * sizeof(float), stream);  // AggS+AggV
    k_edge<<<(tE + 3) / 4, 128, E_LDS, stream>>>(
        Sb, Vb, PF(5), PF(6), PI(7), PI(8), AggS, AggV, PF(b + 0), PF(b + 1),
        PF(b + 2), PF(b + 3), PF(b + 4), PF(b + 5), PF(b + 6), PF(b + 7), PF(b + 8),
        PF(b + 9), tE, E);
    k_node<<<(tN + 3) / 4, 128, N_LDS, stream>>>(
        Sb, Vb, AggS, AggV, PF(3), PF(b + 10), PF(b + 11), PF(b + 12), PF(b + 13),
        PF(b + 14), PF(b + 15), PF(b + 16), PF(b + 17), PF(b + 18), PF(b + 19), tN, N);
  }

  hipMemsetAsync(Pooled, 0, ((size_t)G * 64 + G) * sizeof(float), stream);
  k_pre<<<(tN + 3) / 4, 128, P_LDS, stream>>>(Sb, Vb, PF(3), PI(9), Pooled, PF(55),
                                              PF(56), PF(57), PF(58), PF(59), PF(60),
                                              PF(61), tN, N);
  k_counts<<<(N + 255) / 256, 256, 0, stream>>>(PI(9), Counts, N);
  k_post<<<(G + 127) / 128, 128, 0, stream>>>(Pooled, Counts, PF(62), PF(63),
                                              (float*)d_out, G);
}
